// Attention_59459527246143
// MI455X (gfx1250) — matrix operand order repaired and checked by reading, not compiled, not run on hardware
//
#include <hip/hip_runtime.h>

#define Hdim 1024
#define Bsz  8
#define Ssz  4096
#define M_TOTAL (Bsz * Ssz)      // 32768 rows
#define K2 (2 * Hdim)            // 2048 combined K
#define M_TILE 64
#define LDS_PITCH (K2 + 8)       // 2056 halves: +16B pad -> conflict-free b128

typedef __bf16 bf16_t;
typedef __attribute__((ext_vector_type(16))) __bf16 v16bf;
typedef __attribute__((ext_vector_type(8)))  __bf16 v8bf;
typedef __attribute__((ext_vector_type(4)))  __bf16 v4bf;
typedef __attribute__((ext_vector_type(8)))  float  v8f;
typedef __attribute__((ext_vector_type(4)))  float  v4f;

union BF16x16 { v16bf v; v8bf h[2]; };

#define WMMA_BF16(A, B, C) \
    __builtin_amdgcn_wmma_f32_16x16x32_bf16(false, (A).v, false, (B).v, (short)0, (C), false, false)

// ---------------------------------------------------------------- kernel 0:
// Pre-convert [Qw | Kw] (row n, cols k=0..2047) to bf16 in workspace; also
// combined bias Qb+Kb. Runs once per launch; ~12MB of traffic, negligible.
__global__ __launch_bounds__(256) void convert_weights(
    const float* __restrict__ Qw, const float* __restrict__ Kw,
    const float* __restrict__ Qb, const float* __restrict__ Kb,
    bf16_t* __restrict__ Wbf, float* __restrict__ biasC)
{
    int tid  = blockIdx.x * blockDim.x + threadIdx.x;
    int base = tid * 8;
    #pragma unroll
    for (int i = 0; i < 8; ++i) {
        int idx = base + i;
        int n = idx >> 11;          // / K2
        int k = idx & (K2 - 1);
        float x = (k < Hdim) ? Qw[n * Hdim + k] : Kw[n * Hdim + (k - Hdim)];
        Wbf[idx] = (bf16_t)x;
    }
    if (tid < Hdim) biasC[tid] = Qb[tid] + Kb[tid];
}

// ---------------------------------------------------------------- kernel 1:
// Fused dual-GEMM + bias + tanh + Vw-dot -> raw scores.
// Block: 256 threads (8 waves). Block owns 64 rows of [queries‖keys] in LDS
// (bf16, ~257KB). Wave w: M-pair (w>>2) -> rows (w>>2)*32..+31; N-set (w&3)
// -> 16 N-tiles processed two at a time. 2x2 register blocking: per K-step
// load a0,a1,b0,b1 once and issue 4 WMMAs -> 1 ds + 1 global b128 per WMMA.
__global__ __launch_bounds__(256) void scores_kernel(
    const float* __restrict__ queries, const float* __restrict__ keys,
    const bf16_t* __restrict__ Wbf, const float* __restrict__ biasC,
    const float* __restrict__ Vw,  const float* __restrict__ Vb,
    float* __restrict__ scoresOut)
{
    __shared__ __align__(16) bf16_t tileA[M_TILE * LDS_PITCH];
    __shared__ float sBias[Hdim];
    __shared__ float sVw[Hdim];
    __shared__ float sPart[8 * 32];

    const int tid     = threadIdx.x;
    const int rowBase = blockIdx.x * M_TILE;

    // Load A tile (64 x 2048 f32, b128-coalesced) -> bf16 LDS.
    // queries: read-once stream -> non-temporal (keep keys+weights in L2).
    for (int idx = tid; idx < M_TILE * K2 / 4; idx += 256) {
        int e = idx * 4;
        int r = e >> 11;
        int c = e & (K2 - 1);
        int g = rowBase + r;
        v4f x;
        if (c < Hdim)
            x = __builtin_nontemporal_load((const v4f*)(queries + (size_t)g * Hdim + c));
        else
            x = *(const v4f*)(keys + (size_t)g * Hdim + (c - Hdim));
        v4bf y;
        #pragma unroll
        for (int j = 0; j < 4; ++j) y[j] = (bf16_t)x[j];
        *(v4bf*)(&tileA[r * LDS_PITCH + c]) = y;
    }
    for (int i = tid; i < Hdim; i += 256) { sBias[i] = biasC[i]; sVw[i] = Vw[i]; }
    __syncthreads();

    const int w    = tid >> 5;
    const int lane = tid & 31;
    const int ln   = lane & 15;
    const int hi   = lane >> 4;      // lane half selects K sub-range
    const int mPair = w >> 2;        // rows mPair*32 .. +31 (two 16-row tiles)
    const int nSet  = w & 3;         // 16 N-tiles, done in pairs

    float partial0[8], partial1[8];
    #pragma unroll
    for (int i = 0; i < 8; ++i) { partial0[i] = 0.f; partial1[i] = 0.f; }

    // A-matrix 16x32 bf16 layout: lane = M row (both halves);
    // halves 0-7 -> K = base+0..7 (+8 if lane>=16), halves 8-15 -> +16.
    const bf16_t* aRow0 = &tileA[(mPair * 32 + ln) * LDS_PITCH];
    const bf16_t* aRow1 = aRow0 + 16 * LDS_PITCH;

    for (int t = 0; t < 8; ++t) {
        const int nt0 = nSet * 16 + 2 * t;
        const int n0  = nt0 * 16 + ln;     // this lane's column in tile pair
        // B-matrix 32x16: lane = N col; halves 0-7 -> K 0..7 (+8 if lane>=16), halves 8-15 -> +16
        const bf16_t* bRow0 = &Wbf[(size_t)n0 * K2 + hi * 8];
        const bf16_t* bRow1 = bRow0 + (size_t)16 * K2;   // next N-tile (n0+16)

        v8f acc00 = {}, acc01 = {}, acc10 = {}, acc11 = {};
        #pragma unroll 2
        for (int kk = 0; kk < K2; kk += 32) {
            BF16x16 a0, a1, b0, b1;
            a0.h[0] = *(const v8bf*)(aRow0 + kk + hi * 8);        // ds_load_b128
            a0.h[1] = *(const v8bf*)(aRow0 + kk + 16 + hi * 8);
            a1.h[0] = *(const v8bf*)(aRow1 + kk + hi * 8);
            a1.h[1] = *(const v8bf*)(aRow1 + kk + 16 + hi * 8);
            b0.h[0] = *(const v8bf*)(bRow0 + kk);                 // global_load_b128
            b0.h[1] = *(const v8bf*)(bRow0 + kk + 16);
            b1.h[0] = *(const v8bf*)(bRow1 + kk);
            b1.h[1] = *(const v8bf*)(bRow1 + kk + 16);
            acc00 = WMMA_BF16(a0, b0, acc00);
            acc01 = WMMA_BF16(a0, b1, acc01);
            acc10 = WMMA_BF16(a1, b0, acc10);
            acc11 = WMMA_BF16(a1, b1, acc11);
        }
        // Epilogue: C/D layout -> lane ln is column, VGPR i is row i+8*hi
        const float bias0 = sBias[n0],      vw0 = sVw[n0];
        const float bias1 = sBias[n0 + 16], vw1 = sVw[n0 + 16];
        #pragma unroll
        for (int i = 0; i < 8; ++i) {
            partial0[i] += tanhf(acc00[i] + bias0) * vw0 + tanhf(acc01[i] + bias1) * vw1;
            partial1[i] += tanhf(acc10[i] + bias0) * vw0 + tanhf(acc11[i] + bias1) * vw1;
        }
    }

    // Reduce the n-partials across the 16 lanes of each half
    #pragma unroll
    for (int i = 0; i < 8; ++i) {
        float p0 = partial0[i], p1 = partial1[i];
        p0 += __shfl_xor(p0, 1, 32);  p1 += __shfl_xor(p1, 1, 32);
        p0 += __shfl_xor(p0, 2, 32);  p1 += __shfl_xor(p1, 2, 32);
        p0 += __shfl_xor(p0, 4, 32);  p1 += __shfl_xor(p1, 4, 32);
        p0 += __shfl_xor(p0, 8, 32);  p1 += __shfl_xor(p1, 8, 32);
        if (ln == 0) {
            sPart[w * 32 +      hi * 8 + i] = p0;   // rows mPair*32 + 0..15
            sPart[w * 32 + 16 + hi * 8 + i] = p1;   // rows mPair*32 + 16..31
        }
    }
    __syncthreads();

    // Cross-wave (4 N-sets) reduce, add Vb, emit raw score
    if (tid < M_TILE) {
        int mp = tid >> 5, lrow = tid & 31;
        float s = 0.f;
        #pragma unroll
        for (int j = 0; j < 4; ++j) s += sPart[(mp * 4 + j) * 32 + lrow];
        scoresOut[rowBase + tid] = s + Vb[0];
    }
}

// ---------------------------------------------------------------- kernel 2:
// In-place softmax over S=4096 per batch. One block per batch.
__global__ __launch_bounds__(256) void softmax_kernel(float* __restrict__ weights)
{
    __shared__ float red[256];
    float* wsrow = weights + blockIdx.x * Ssz;
    const int tid = threadIdx.x;

    float e[16];
    float mx = -3.4e38f;
    #pragma unroll
    for (int i = 0; i < 16; ++i) { e[i] = wsrow[tid + i * 256]; mx = fmaxf(mx, e[i]); }
    red[tid] = mx; __syncthreads();
    for (int off = 128; off > 0; off >>= 1) {
        if (tid < off) red[tid] = fmaxf(red[tid], red[tid + off]);
        __syncthreads();
    }
    mx = red[0]; __syncthreads();

    float sum = 0.f;
    #pragma unroll
    for (int i = 0; i < 16; ++i) { e[i] = __expf(e[i] - mx); sum += e[i]; }
    red[tid] = sum; __syncthreads();
    for (int off = 128; off > 0; off >>= 1) {
        if (tid < off) red[tid] += red[tid + off];
        __syncthreads();
    }
    const float inv = 1.0f / red[0];
    #pragma unroll
    for (int i = 0; i < 16; ++i) wsrow[tid + i * 256] = e[i] * inv;
}

// ---------------------------------------------------------------- kernel 3/4:
__global__ void zero_ctx(float* ctx)
{
    ctx[blockIdx.x * blockDim.x + threadIdx.x] = 0.f;
}

// context[b,h] = sum_s weights[b,s]*keys[b,s,h]; split S across blocks,
// coalesced over h, float atomics to accumulate partials. keys should be
// mostly L2-resident (queries were streamed non-temporally in kernel 1).
__global__ __launch_bounds__(256) void context_kernel(
    const float* __restrict__ keys, const float* __restrict__ weights,
    float* __restrict__ ctx)
{
    const int b  = blockIdx.z;
    const int sc = blockIdx.y;                 // 0..15 (chunks of 256 s)
    const int h  = blockIdx.x * 256 + threadIdx.x;
    const float* wp = weights + b * Ssz + sc * 256;
    const float* kp = keys + ((size_t)b * Ssz + sc * 256) * Hdim + h;
    float acc = 0.f;
    #pragma unroll 4
    for (int s = 0; s < 256; ++s) acc += wp[s] * kp[(size_t)s * Hdim];
    unsafeAtomicAdd(&ctx[b * Hdim + h], acc);  // global_atomic_add_f32
}

// ----------------------------------------------------------------
extern "C" void kernel_launch(void* const* d_in, const int* in_sizes, int n_in,
                              void* d_out, int out_size, void* d_ws, size_t ws_size,
                              hipStream_t stream)
{
    const float* queries = (const float*)d_in[0];
    const float* keys    = (const float*)d_in[1];
    const float* Qw      = (const float*)d_in[2];
    const float* Qb      = (const float*)d_in[3];
    const float* Kw      = (const float*)d_in[4];
    const float* Kb      = (const float*)d_in[5];
    const float* Vw      = (const float*)d_in[6];
    const float* Vb      = (const float*)d_in[7];

    float* outCtx = (float*)d_out;                 // [8,1,1024] = 8192 floats
    float* outWts = (float*)d_out + Bsz * Hdim;    // [8,1,4096] = 32768 floats

    bf16_t* Wbf  = (bf16_t*)d_ws;                                      // 4 MB
    float*  bias = (float*)((char*)d_ws + (size_t)Hdim * K2 * sizeof(bf16_t));

    // 1) weights -> bf16 (+ combined bias)
    convert_weights<<<(Hdim * K2 / 8) / 256, 256, 0, stream>>>(Qw, Kw, Qb, Kb, Wbf, bias);

    // 2) fused GEMM+tanh+dot -> raw scores (into weights slot of d_out)
    scores_kernel<<<M_TOTAL / M_TILE, 256, 0, stream>>>(
        queries, keys, Wbf, bias, Vw, Vb, outWts);

    // 3) softmax in place
    softmax_kernel<<<Bsz, 256, 0, stream>>>(outWts);

    // 4) context = weights @ keys
    zero_ctx<<<(Bsz * Hdim) / 256, 256, 0, stream>>>(outCtx);
    context_kernel<<<dim3(Hdim / 256, Ssz / 256, Bsz), 256, 0, stream>>>(
        keys, outWts, outCtx);
}
